// Attention_85057532330254
// MI455X (gfx1250) — matrix operand order repaired and checked by reading, not compiled, not run on hardware
//
#include <hip/hip_runtime.h>
#include <hip/hip_bf16.h>

// ---------------------------------------------------------------- constants
#define BATCH   16
#define NTOK    1024            // 32*32
#define ROWS    16384           // BATCH*NTOK
#define DIM     256
#define HEADS   8
#define DK      32
#define DV      64
#define INNER_K 256             // HEADS*DK
#define INNER_V 512             // HEADS*DV
#define QKV_N   1024            // INNER_K*2 + INNER_V
#define EPSBN   1e-5f

typedef __attribute__((ext_vector_type(16))) __bf16 v16bf;
typedef __attribute__((ext_vector_type(8)))  float  v8f;

__device__ __forceinline__ unsigned short f2bf(float f) {
    unsigned u = __float_as_uint(f);
    unsigned r = u + 0x7FFFu + ((u >> 16) & 1u);   // round-to-nearest-even
    return (unsigned short)(r >> 16);
}

// ---------------------------------------------------------------- converters
__global__ void cvt_bf16_kernel(const float* __restrict__ src,
                                unsigned short* __restrict__ dst, size_t n) {
    size_t i = (size_t)blockIdx.x * blockDim.x + threadIdx.x;
    if (i < n) dst[i] = f2bf(src[i]);
}

// WqkvT[n][k] = {wq|wk|wv}(k, n)   (transposed so B-operand loads are contiguous)
__global__ void cvt_wqkvT_kernel(const float* __restrict__ wq,
                                 const float* __restrict__ wk,
                                 const float* __restrict__ wv,
                                 unsigned short* __restrict__ wT) {
    int id = blockIdx.x * 256 + threadIdx.x;      // QKV_N * DIM
    int n = id >> 8, k = id & 255;
    float v;
    if (n < 256)       v = wq[k * 256 + n];
    else if (n < 512)  v = wk[k * 256 + (n - 256)];
    else               v = wv[k * 512 + (n - 512)];
    wT[n * DIM + k] = f2bf(v);
}

// WoutT[n][k] = w_out(k, n),  n<256, k<512
__global__ void cvt_woutT_kernel(const float* __restrict__ w,
                                 unsigned short* __restrict__ wT) {
    int id = blockIdx.x * 256 + threadIdx.x;      // 256*512
    int n = id >> 9, k = id & 511;
    wT[n * INNER_V + k] = f2bf(w[k * 256 + n]);
}

__global__ void zerof_kernel(float* p, int n) {
    int i = blockIdx.x * blockDim.x + threadIdx.x;
    if (i < n) p[i] = 0.f;
}

// ------------------------------------------------- GEMM + BN-stats (WMMA bf16)
// P[M,N] = A[M,K](bf16) * W[K,N](bf16, stored transposed WT[N,K]) + bias[N]
// Also accumulates per-column sum / sumsq for BatchNorm statistics.
// block = 256 thr (8 waves, 2 row-tiles x 4 col-tiles), wave tile = 16x16.
__global__ void __launch_bounds__(256)
gemm_bn_kernel(const unsigned short* __restrict__ A, int lda,
               const unsigned short* __restrict__ WT, int ldw,
               const float* __restrict__ bias,
               float* __restrict__ P, float* __restrict__ s1,
               float* __restrict__ s2, int N, int K) {
    int lane = threadIdx.x & 31, wv = threadIdx.x >> 5;
    int rt = wv >> 2, ct = wv & 3;
    int row0 = blockIdx.y * 32 + rt * 16;
    int col0 = blockIdx.x * 64 + ct * 16;
    int half = lane >> 4, mn = lane & 15;

    v8f acc = {};
    const unsigned short* arow = A  + (size_t)(row0 + mn) * lda;
    const unsigned short* wrow = WT + (size_t)(col0 + mn) * ldw;

    for (int k0 = 0; k0 < K; k0 += 32) {
        v16bf a, b;
        // A layout: slot s -> k = half*16 + s  (16 contiguous bf16)
        *(uint4*)&a         = *(const uint4*)(arow + k0 + half * 16);
        *(((uint4*)&a) + 1) = *(const uint4*)(arow + k0 + half * 16 + 8);
        // B layout: slot s -> k = half*16 + s  (16 contiguous bf16)
        *(uint4*)&b         = *(const uint4*)(wrow + k0 + half * 16);
        *(((uint4*)&b) + 1) = *(const uint4*)(wrow + k0 + half * 16 + 8);
        acc = __builtin_amdgcn_wmma_f32_16x16x32_bf16(false, a, false, b,
                                                      (short)0, acc, false, false);
    }

    int col = col0 + mn;
    float bval = bias ? bias[col] : 0.f;
    float csum = 0.f, csq = 0.f;
#pragma unroll
    for (int v = 0; v < 8; ++v) {
        int m = v + 8 * half;                 // C layout: row = v + 8*(lane>>4)
        float val = acc[v] + bval;
        P[(size_t)(row0 + m) * N + col] = val;
        csum += val; csq += val * val;
    }
    csum += __shfl_xor(csum, 16, 32);
    csq  += __shfl_xor(csq, 16, 32);
    if (half == 0) { atomicAdd(&s1[col], csum); atomicAdd(&s2[col], csq); }
}

// -------------------------------------------- BN finalize: a = g*rsqrt(var+e)
__global__ void bn_finalize_kernel(const float* s1, const float* s2,
                                   const float* g0, const float* b0, int n0,
                                   const float* g1, const float* b1, int n1,
                                   const float* g2, const float* b2,
                                   float* a, float* c, int ncols, float invM) {
    int j = blockIdx.x * blockDim.x + threadIdx.x;
    if (j >= ncols) return;
    float mean = s1[j] * invM;
    float var  = s2[j] * invM - mean * mean;
    float g, bb;
    if (j < n0)           { g = g0[j];           bb = b0[j]; }
    else if (j < n0 + n1) { g = g1[j - n0];      bb = b1[j - n0]; }
    else                  { g = g2[j - n0 - n1]; bb = b2[j - n0 - n1]; }
    float av = g * rsqrtf(var + EPSBN);
    a[j] = av; c[j] = bb - mean * av;
}

// ------------------------- normalize + repack: q,k row-major, V transposed
__global__ void normalize_qkv_kernel(const float* __restrict__ P,
                                     const float* __restrict__ a,
                                     const float* __restrict__ c,
                                     unsigned short* __restrict__ qn,
                                     unsigned short* __restrict__ kn,
                                     unsigned short* __restrict__ vT) {
    size_t id = (size_t)blockIdx.x * blockDim.x + threadIdx.x;   // ROWS*QKV_N
    int col = (int)(id & 1023);
    size_t row = id >> 10;
    unsigned short bf = f2bf(P[id] * a[col] + c[col]);
    if (col < 256)      qn[row * 256 + col] = bf;
    else if (col < 512) kn[row * 256 + (col - 256)] = bf;
    else {
        int b = (int)(row >> 10), n = (int)(row & 1023);
        int cc = col - 512, h = cc >> 6, d = cc & 63;
        vT[(((size_t)(b * 8 + h) * 64 + d) << 10) + n] = bf;
    }
}

// ------------------------------------------------ flash attention (WMMA bf16)
// one wave per (batch, head, 16-query tile); online softmax over 32-key steps
__global__ void __launch_bounds__(128)
attn_kernel(const unsigned short* __restrict__ qn,
            const unsigned short* __restrict__ kn,
            const unsigned short* __restrict__ vT,
            const float* __restrict__ pos_emb,
            unsigned short* __restrict__ attnb) {
    __shared__ __align__(32) unsigned short lds_p[4][16 * 32];

    int lane = threadIdx.x & 31, w = threadIdx.x >> 5;
    int gid = blockIdx.x * 4 + w;
    int b  = gid >> 9;            // 8 heads * 64 tiles = 512 per batch
    int h  = (gid >> 6) & 7;
    int rt = gid & 63;
    int half = lane >> 4, mn = lane & 15;

    // Q tile in A layout (per-lane: two 16B contiguous chunks)
    const unsigned short* qptr =
        qn + (size_t)(b * NTOK + rt * 16 + mn) * INNER_K + h * DK;
    v16bf aq;
    *(uint4*)&aq         = *(const uint4*)(qptr + half * 16);
    *(((uint4*)&aq) + 1) = *(const uint4*)(qptr + half * 16 + 8);

    v8f O0 = {}, O1 = {}, O2 = {}, O3 = {};
    float rmax[8], rsum[8];
#pragma unroll
    for (int v = 0; v < 8; ++v) { rmax[v] = -1e30f; rsum[v] = 0.f; }

    const float scale = 0.17677669529f;        // 32^-0.5
    const float invscale = 5.65685424949f;     // 32^0.5
    const unsigned short* kbase = kn + (size_t)b * NTOK * INNER_K + h * DK;
    const unsigned short* vbase = vT + ((size_t)(b * 8 + h) * 64) * NTOK;
    unsigned short* lp = lds_p[w];

    int i_lane = rt * 16;                      // query-tile base
    for (int jt = 0; jt < 32; ++jt) {
        int j0 = jt * 32;
        // K tiles as B operand: lane n = mn, kk = half*16 + s (contiguous)
        v16bf bk0, bk1;
        {
            const unsigned short* p0 = kbase + (size_t)(j0 + mn) * INNER_K + half * 16;
            *(uint4*)&bk0         = *(const uint4*)p0;
            *(((uint4*)&bk0) + 1) = *(const uint4*)(p0 + 8);
            const unsigned short* p1 = kbase + (size_t)(j0 + 16 + mn) * INNER_K + half * 16;
            *(uint4*)&bk1         = *(const uint4*)p1;
            *(((uint4*)&bk1) + 1) = *(const uint4*)(p1 + 8);
        }
        v8f S0 = {}, S1 = {};
        S0 = __builtin_amdgcn_wmma_f32_16x16x32_bf16(false, aq, false, bk0,
                                                     (short)0, S0, false, false);
        S1 = __builtin_amdgcn_wmma_f32_16x16x32_bf16(false, aq, false, bk1,
                                                     (short)0, S1, false, false);

        float p0v[8], p1v[8];
#pragma unroll
        for (int v = 0; v < 8; ++v) {
            int m = v + 8 * half;
            int i = i_lane + m;
            int xi = i >> 5, yi = i & 31;
            int ja = j0 + mn, jb = j0 + 16 + mn;
            int dx0 = xi - (ja >> 5); dx0 = dx0 < 0 ? -dx0 : dx0;
            int dy0 = yi - (ja & 31); dy0 = dy0 < 0 ? -dy0 : dy0;
            int dx1 = xi - (jb >> 5); dx1 = dx1 < 0 ? -dx1 : dx1;
            int dy1 = yi - (jb & 31); dy1 = dy1 < 0 ? -dy1 : dy1;
            p0v[v] = S0[v] * scale + pos_emb[(dx0 * 32 + dy0) * 8 + h] * invscale;
            p1v[v] = S1[v] * scale + pos_emb[(dx1 * 32 + dy1) * 8 + h] * invscale;
        }
        // online softmax: row reductions across the 16-lane half-groups
#pragma unroll
        for (int v = 0; v < 8; ++v) {
            float mx = fmaxf(p0v[v], p1v[v]);
            mx = fmaxf(mx, __shfl_xor(mx, 1, 32));
            mx = fmaxf(mx, __shfl_xor(mx, 2, 32));
            mx = fmaxf(mx, __shfl_xor(mx, 4, 32));
            mx = fmaxf(mx, __shfl_xor(mx, 8, 32));
            float nm = fmaxf(rmax[v], mx);
            float alpha = __expf(rmax[v] - nm);
            float e0 = __expf(p0v[v] - nm), e1 = __expf(p1v[v] - nm);
            float ps = e0 + e1;
            ps += __shfl_xor(ps, 1, 32);
            ps += __shfl_xor(ps, 2, 32);
            ps += __shfl_xor(ps, 4, 32);
            ps += __shfl_xor(ps, 8, 32);
            rsum[v] = rsum[v] * alpha + ps;
            rmax[v] = nm;
            O0[v] *= alpha; O1[v] *= alpha; O2[v] *= alpha; O3[v] *= alpha;
            p0v[v] = e0; p1v[v] = e1;
        }
        // bounce probabilities through LDS into the 16-bit A register layout
#pragma unroll
        for (int v = 0; v < 8; ++v) {
            int m  = v + 8 * half;
            int k0c = mn, k1c = 16 + mn;
            lp[((k0c >> 4) * 16 + m) * 16 + (k0c & 15)] = f2bf(p0v[v]);
            lp[((k1c >> 4) * 16 + m) * 16 + (k1c & 15)] = f2bf(p1v[v]);
        }
        asm volatile("s_wait_dscnt 0" ::: "memory");
        v16bf pa = *(const v16bf*)(lp + lane * 16);

        // V tiles (vT layout -> contiguous B-operand loads): O_t += P * V_t
        const unsigned short* vb = vbase + j0 + half * 16;
        v16bf bv;
        const unsigned short* pv0 = vb + (size_t)(0 * 16 + mn) * NTOK;
        *(uint4*)&bv = *(const uint4*)pv0; *(((uint4*)&bv) + 1) = *(const uint4*)(pv0 + 8);
        O0 = __builtin_amdgcn_wmma_f32_16x16x32_bf16(false, pa, false, bv, (short)0, O0, false, false);
        const unsigned short* pv1 = vb + (size_t)(1 * 16 + mn) * NTOK;
        *(uint4*)&bv = *(const uint4*)pv1; *(((uint4*)&bv) + 1) = *(const uint4*)(pv1 + 8);
        O1 = __builtin_amdgcn_wmma_f32_16x16x32_bf16(false, pa, false, bv, (short)0, O1, false, false);
        const unsigned short* pv2 = vb + (size_t)(2 * 16 + mn) * NTOK;
        *(uint4*)&bv = *(const uint4*)pv2; *(((uint4*)&bv) + 1) = *(const uint4*)(pv2 + 8);
        O2 = __builtin_amdgcn_wmma_f32_16x16x32_bf16(false, pa, false, bv, (short)0, O2, false, false);
        const unsigned short* pv3 = vb + (size_t)(3 * 16 + mn) * NTOK;
        *(uint4*)&bv = *(const uint4*)pv3; *(((uint4*)&bv) + 1) = *(const uint4*)(pv3 + 8);
        O3 = __builtin_amdgcn_wmma_f32_16x16x32_bf16(false, pa, false, bv, (short)0, O3, false, false);
    }

    // epilogue: 1/rowsum, tanh-GELU, write bf16 [row, h*64 + c]
#pragma unroll
    for (int v = 0; v < 8; ++v) {
        int m = v + 8 * half;
        size_t row = (size_t)(b * NTOK + i_lane + m);
        float ir = 1.0f / rsum[v];
        float o[4] = { O0[v] * ir, O1[v] * ir, O2[v] * ir, O3[v] * ir };
#pragma unroll
        for (int t = 0; t < 4; ++t) {
            float x = o[t];
            float g = 0.5f * x * (1.0f + tanhf(0.7978845608f * (x + 0.044715f * x * x * x)));
            attnb[row * INNER_V + h * DV + t * 16 + mn] = f2bf(g);
        }
    }
}

// ---------------------------------------------------------- final normalize
__global__ void final_norm_kernel(const float* __restrict__ P,
                                  const float* __restrict__ a,
                                  const float* __restrict__ c,
                                  float* __restrict__ out) {
    size_t id = (size_t)blockIdx.x * blockDim.x + threadIdx.x;  // ROWS*256
    int col = (int)(id & 255);
    out[id] = P[id] * a[col] + c[col];
}

// ---------------------------------------------------------------- workspace
static constexpr size_t XB_OFF    = 0;                        // 16384*256*2
static constexpr size_t WQKVT_OFF = XB_OFF    + 8388608;      // 1024*256*2
static constexpr size_t WOUTT_OFF = WQKVT_OFF + 524288;       // 256*512*2
static constexpr size_t P_OFF     = WOUTT_OFF + 262144;       // 16384*1024*4 (reused as P2)
static constexpr size_t STAT_OFF  = P_OFF     + 67108864;     // 2*1024*4 (sum|sumsq, reused)
static constexpr size_t AC_OFF    = STAT_OFF  + 8192;         // 2*1024*4 (a|c, reused)
static constexpr size_t QN_OFF    = AC_OFF    + 8192;         // 16384*256*2
static constexpr size_t KN_OFF    = QN_OFF    + 8388608;      // 16384*256*2
static constexpr size_t VT_OFF    = KN_OFF    + 8388608;      // 16*8*64*1024*2
static constexpr size_t ATTN_OFF  = VT_OFF    + 16777216;     // 16384*512*2

extern "C" void kernel_launch(void* const* d_in, const int* in_sizes, int n_in,
                              void* d_out, int out_size, void* d_ws, size_t ws_size,
                              hipStream_t stream) {
    const float* x       = (const float*)d_in[0];
    const float* wq      = (const float*)d_in[1];
    const float* gq      = (const float*)d_in[2];
    const float* bq      = (const float*)d_in[3];
    const float* wk      = (const float*)d_in[4];
    const float* gk      = (const float*)d_in[5];
    const float* bk      = (const float*)d_in[6];
    const float* wv      = (const float*)d_in[7];
    const float* gv      = (const float*)d_in[8];
    const float* bv      = (const float*)d_in[9];
    const float* pos_emb = (const float*)d_in[10];
    const float* w_out   = (const float*)d_in[11];
    const float* b_out   = (const float*)d_in[12];
    const float* go      = (const float*)d_in[13];
    const float* bo      = (const float*)d_in[14];
    float* out = (float*)d_out;

    char* ws = (char*)d_ws;
    unsigned short* xb    = (unsigned short*)(ws + XB_OFF);
    unsigned short* wqkvT = (unsigned short*)(ws + WQKVT_OFF);
    unsigned short* woutT = (unsigned short*)(ws + WOUTT_OFF);
    float*          P     = (float*)(ws + P_OFF);      // also P2
    float*          s1    = (float*)(ws + STAT_OFF);
    float*          s2    = s1 + 1024;
    float*          an    = (float*)(ws + AC_OFF);
    float*          cn    = an + 1024;
    unsigned short* qn    = (unsigned short*)(ws + QN_OFF);
    unsigned short* kn    = (unsigned short*)(ws + KN_OFF);
    unsigned short* vT    = (unsigned short*)(ws + VT_OFF);
    unsigned short* attnb = (unsigned short*)(ws + ATTN_OFF);

    // 1) bf16 conversions (x, transposed weights)
    cvt_bf16_kernel<<<(ROWS * DIM) / 256, 256, 0, stream>>>(x, xb, (size_t)ROWS * DIM);
    cvt_wqkvT_kernel<<<(QKV_N * DIM) / 256, 256, 0, stream>>>(wq, wk, wv, wqkvT);
    cvt_woutT_kernel<<<(DIM * INNER_V) / 256, 256, 0, stream>>>(w_out, woutT);

    // 2) QKV projection GEMM + BN stats
    zerof_kernel<<<8, 256, 0, stream>>>(s1, 2048);
    gemm_bn_kernel<<<dim3(QKV_N / 64, ROWS / 32), 256, 0, stream>>>(
        xb, DIM, wqkvT, DIM, nullptr, P, s1, s2, QKV_N, DIM);
    bn_finalize_kernel<<<4, 256, 0, stream>>>(s1, s2, gq, bq, 256, gk, bk, 256,
                                              gv, bv, an, cn, QKV_N, 1.0f / ROWS);

    // 3) normalize + repack (q,k row-major bf16; V transposed per head)
    normalize_qkv_kernel<<<(size_t)ROWS * QKV_N / 256, 256, 0, stream>>>(
        P, an, cn, qn, kn, vT);

    // 4) flash attention + GELU (bf16 WMMA throughout)
    attn_kernel<<<(BATCH * HEADS * 64) / 4, 128, 0, stream>>>(qn, kn, vT, pos_emb, attnb);

    // 5) output projection GEMM (+bias) + BN stats
    zerof_kernel<<<8, 256, 0, stream>>>(s1, 2048);
    gemm_bn_kernel<<<dim3(DIM / 64, ROWS / 32), 256, 0, stream>>>(
        attnb, INNER_V, woutT, INNER_V, b_out, P, s1, s2, DIM, INNER_V);
    bn_finalize_kernel<<<1, 256, 0, stream>>>(s1, s2, go, bo, 256, go, bo, 0,
                                              go, bo, an, cn, DIM, 1.0f / ROWS);

    // 6) final BatchNorm apply -> f32 output
    final_norm_kernel<<<(ROWS * DIM) / 256, 256, 0, stream>>>(P, an, cn, out);
}
